// ZigzagAttention_24592982737555
// MI455X (gfx1250) — compile-verified, matrix operand order checked by reading, not run on hardware
//
#include <hip/hip_runtime.h>
#include <cstdint>

typedef __attribute__((ext_vector_type(16))) __bf16 v16bf;
typedef __attribute__((ext_vector_type(8)))  float  v8f;

#define B_  2
#define S_  4096
#define D_  1024
#define H_  16
#define DP_ 64
#define T_  2048

// ---------------- workspace layout (bf16 elements) ----------------
static constexpr size_t XQ_OFF  = 0;
static constexpr size_t XK_OFF  = XQ_OFF + (size_t)B_*S_*D_;
static constexpr size_t XV_OFF  = XK_OFF + (size_t)B_*S_*D_;
static constexpr size_t WQ_OFF  = XV_OFF + (size_t)B_*S_*D_;
static constexpr size_t WK_OFF  = WQ_OFF + (size_t)D_*D_;
static constexpr size_t WV_OFF  = WK_OFF + (size_t)D_*D_;
static constexpr size_t WFC_OFF = WV_OFF + (size_t)D_*D_;
static constexpr size_t QH_OFF  = WFC_OFF + (size_t)D_*D_;
static constexpr size_t KH_OFF  = QH_OFF + (size_t)B_*S_*D_;
static constexpr size_t VH_OFF  = KH_OFF + (size_t)B_*S_*D_;
static constexpr size_t AO_OFF  = VH_OFF + (size_t)B_*S_*D_;

static constexpr size_t OUT_ELEMS = (size_t)B_*S_*D_;          // 8,388,608
static constexpr size_t W_SPLIT   = (size_t)B_*H_*T_*T_;       // 134,217,728

// ---------------- CDNA5 helpers ----------------
__device__ __forceinline__ unsigned lds_addr(const void* p) {
  return (unsigned)(uintptr_t)p;   // low 32 bits of LDS-aperture flat address
}

// async global -> LDS, 16 bytes (tracked by ASYNCcnt)
__device__ __forceinline__ void async_b128(unsigned lds, const void* g) {
  asm volatile("global_load_async_to_lds_b128 %0, %1, off"
               :: "v"(lds), "v"((unsigned long long)(uintptr_t)g)
               : "memory");
}
__device__ __forceinline__ void wait_async0() {
  asm volatile("s_wait_asynccnt 0x0" ::: "memory");
}

// A/B fragment load: row = 32 bf16, K-halves per ISA 16-bit A layout
__device__ __forceinline__ v16bf frag_ld(const __bf16* row, int lh) {
  union { v16bf v; uint4 u[2]; } f;
  const uint4* p = (const uint4*)row + lh;   // lh*8 bf16
  f.u[0] = p[0];                              // K = lh*8 .. lh*8+7
  f.u[1] = p[2];                              // K = lh*8+16 .. lh*8+23
  return f.v;
}

__device__ __forceinline__ v8f bf16_wmma(v16bf a, v16bf b, v8f c) {
  return __builtin_amdgcn_wmma_f32_16x16x32_bf16(
      /*neg_a=*/false, a, /*neg_b=*/false, b,
      /*c_mod=*/(short)0, c, /*reuse_a=*/false, /*reuse_b=*/false);
}

// ---------------- fp32 -> bf16 conversion pass ----------------
__global__ __launch_bounds__(256) void cvt_f32_to_bf16(
    const float* __restrict__ s, __bf16* __restrict__ d, int n4) {
  int i = blockIdx.x * 256 + threadIdx.x;
  if (i < n4) {
    const float4 f = ((const float4*)s)[i];
    union { __bf16 h[4]; uint2 u; } o;
    o.h[0] = (__bf16)f.x; o.h[1] = (__bf16)f.y;
    o.h[2] = (__bf16)f.z; o.h[3] = (__bf16)f.w;
    ((uint2*)d)[i] = o.u;
  }
}

// ---------------- Y = X @ W^T + b  (8192x1024x1024, async-LDS + WMMA) -----
// ZIGZAG=true : scatter to [b,h,split,t,dp] bf16 (projection kernels)
// ZIGZAG=false: row-major fp32 (final fc)
template<bool ZIGZAG>
__global__ __launch_bounds__(256) void gemm_xw_kernel(
    const __bf16* __restrict__ X, const __bf16* __restrict__ W,
    const float* __restrict__ bias, void* __restrict__ dst)
{
  __shared__ __bf16 As[128 * 32];
  __shared__ __bf16 Bs[128 * 32];      // N-major: row n holds 32 contiguous K

  const int tid  = threadIdx.x;
  const int lane = tid & 31, wid = tid >> 5;
  const int lh   = lane >> 4, lm = lane & 15;
  const int wm   = wid >> 1,  wn = wid & 1;
  const int m0 = blockIdx.y * 128, n0 = blockIdx.x * 128;

  v8f acc[2][4] = {};

  const int r = tid >> 1, seg = tid & 1;
  const __bf16* Arow = X + (size_t)(m0 + r) * D_ + seg * 16;
  const __bf16* Brow = W + (size_t)(n0 + r) * D_ + seg * 16;
  const unsigned Asd = lds_addr(&As[r * 32 + seg * 16]);
  const unsigned Bsd = lds_addr(&Bs[r * 32 + seg * 16]);

  for (int k0 = 0; k0 < D_; k0 += 32) {
    __syncthreads();
    async_b128(Asd,      Arow + k0);
    async_b128(Asd + 16, Arow + k0 + 8);
    async_b128(Bsd,      Brow + k0);
    async_b128(Bsd + 16, Brow + k0 + 8);
    __builtin_prefetch(Arow + k0 + 32, 0, 0);
    __builtin_prefetch(Brow + k0 + 32, 0, 0);
    wait_async0();
    __syncthreads();

    v16bf af[2], bfr[4];
#pragma unroll
    for (int i = 0; i < 2; ++i) af[i]  = frag_ld(&As[(wm * 32 + i * 16 + lm) * 32], lh);
#pragma unroll
    for (int j = 0; j < 4; ++j) bfr[j] = frag_ld(&Bs[(wn * 64 + j * 16 + lm) * 32], lh);
#pragma unroll
    for (int i = 0; i < 2; ++i)
#pragma unroll
      for (int j = 0; j < 4; ++j)
        acc[i][j] = bf16_wmma(af[i], bfr[j], acc[i][j]);
  }

#pragma unroll
  for (int i = 0; i < 2; ++i)
#pragma unroll
    for (int j = 0; j < 4; ++j)
#pragma unroll
      for (int rr = 0; rr < 8; ++rr) {
        const int mg = m0 + wm * 32 + i * 16 + rr + lh * 8;
        const int ng = n0 + wn * 64 + j * 16 + lm;
        const float v = acc[i][j][rr] + bias[ng];
        if constexpr (ZIGZAG) {
          const int b = mg >> 12, s = mg & (S_ - 1);
          const int split = s & 1, t = s >> 1;
          const int h = ng >> 6,  dp = ng & 63;
          ((__bf16*)dst)[((((size_t)(b * H_ + h) * 2 + split) * T_) + t) * DP_ + dp] = (__bf16)v;
        } else {
          ((float*)dst)[(size_t)mg * D_ + ng] = v;
        }
      }
}

// ---------------- scores = Q @ K^T * 1/8 per (b,h,split) ----------------
__global__ __launch_bounds__(256) void scores_kernel(
    const __bf16* __restrict__ QH, const __bf16* __restrict__ KH,
    float* __restrict__ wbase)
{
  __shared__ __bf16 As[128 * 32];
  __shared__ __bf16 Bs[128 * 32];

  const int tid  = threadIdx.x;
  const int lane = tid & 31, wid = tid >> 5;
  const int lh   = lane >> 4, lm = lane & 15;
  const int wm   = wid >> 1,  wn = wid & 1;
  const int m0 = blockIdx.y * 128, n0 = blockIdx.x * 128;
  const int z  = blockIdx.z;                 // (b*H+h)*2 + split
  const int split = z & 1, bh = z >> 1;

  const __bf16* Q = QH + (size_t)z * T_ * DP_;
  const __bf16* K = KH + (size_t)z * T_ * DP_;
  float* Wp = wbase + (size_t)split * W_SPLIT + (size_t)bh * T_ * T_;

  v8f acc[2][4] = {};

  const int r = tid >> 1, seg = tid & 1;
  const __bf16* Arow = Q + (size_t)(m0 + r) * DP_ + seg * 16;
  const __bf16* Brow = K + (size_t)(n0 + r) * DP_ + seg * 16;
  const unsigned Asd = lds_addr(&As[r * 32 + seg * 16]);
  const unsigned Bsd = lds_addr(&Bs[r * 32 + seg * 16]);

  for (int k0 = 0; k0 < DP_; k0 += 32) {
    __syncthreads();
    async_b128(Asd,      Arow + k0);
    async_b128(Asd + 16, Arow + k0 + 8);
    async_b128(Bsd,      Brow + k0);
    async_b128(Bsd + 16, Brow + k0 + 8);
    wait_async0();
    __syncthreads();

    v16bf af[2], bfr[4];
#pragma unroll
    for (int i = 0; i < 2; ++i) af[i]  = frag_ld(&As[(wm * 32 + i * 16 + lm) * 32], lh);
#pragma unroll
    for (int j = 0; j < 4; ++j) bfr[j] = frag_ld(&Bs[(wn * 64 + j * 16 + lm) * 32], lh);
#pragma unroll
    for (int i = 0; i < 2; ++i)
#pragma unroll
      for (int j = 0; j < 4; ++j)
        acc[i][j] = bf16_wmma(af[i], bfr[j], acc[i][j]);
  }

#pragma unroll
  for (int i = 0; i < 2; ++i)
#pragma unroll
    for (int j = 0; j < 4; ++j)
#pragma unroll
      for (int rr = 0; rr < 8; ++rr) {
        const int mg = m0 + wm * 32 + i * 16 + rr + lh * 8;
        const int ng = n0 + wn * 64 + j * 16 + lm;
        Wp[(size_t)mg * T_ + ng] = acc[i][j][rr] * 0.125f;   // 1/sqrt(64)
      }
}

// ---------------- in-place row softmax over 2048 columns ----------------
__global__ __launch_bounds__(256) void softmax_kernel(float* __restrict__ wbase) {
  float* p = wbase + (size_t)blockIdx.x * T_;
  const int tid = threadIdx.x;
  float4 a = ((float4*)p)[tid * 2];
  float4 b = ((float4*)p)[tid * 2 + 1];

  float m = fmaxf(fmaxf(fmaxf(a.x, a.y), fmaxf(a.z, a.w)),
                  fmaxf(fmaxf(b.x, b.y), fmaxf(b.z, b.w)));
#pragma unroll
  for (int off = 16; off; off >>= 1) m = fmaxf(m, __shfl_xor(m, off, 32));

  __shared__ float red[8], red2[8];
  const int wid = tid >> 5, lane = tid & 31;
  if (lane == 0) red[wid] = m;
  __syncthreads();
  if (tid == 0) {
    float mm = red[0];
#pragma unroll
    for (int i = 1; i < 8; ++i) mm = fmaxf(mm, red[i]);
    red[0] = mm;
  }
  __syncthreads();
  m = red[0];

  a.x = __expf(a.x - m); a.y = __expf(a.y - m); a.z = __expf(a.z - m); a.w = __expf(a.w - m);
  b.x = __expf(b.x - m); b.y = __expf(b.y - m); b.z = __expf(b.z - m); b.w = __expf(b.w - m);
  float s = a.x + a.y + a.z + a.w + b.x + b.y + b.z + b.w;
#pragma unroll
  for (int off = 16; off; off >>= 1) s += __shfl_xor(s, off, 32);
  if (lane == 0) red2[wid] = s;
  __syncthreads();
  if (tid == 0) {
    float ss = 0.f;
#pragma unroll
    for (int i = 0; i < 8; ++i) ss += red2[i];
    red2[0] = ss;
  }
  __syncthreads();
  const float inv = 1.0f / red2[0];

  a.x *= inv; a.y *= inv; a.z *= inv; a.w *= inv;
  b.x *= inv; b.y *= inv; b.z *= inv; b.w *= inv;
  ((float4*)p)[tid * 2]     = a;
  ((float4*)p)[tid * 2 + 1] = b;
}

// ---------------- O = W @ V per (b,h,split); merged-head bf16 out --------
__global__ __launch_bounds__(256) void wv_kernel(
    const float* __restrict__ wbase, const __bf16* __restrict__ VH,
    __bf16* __restrict__ AO)
{
  __shared__ __bf16 As[128 * 32];
  __shared__ __bf16 Bt[64 * 40];        // [n=dp][k=t], stride 40 (16B aligned)

  const int tid  = threadIdx.x;
  const int lane = tid & 31, wid = tid >> 5;
  const int lh   = lane >> 4, lm = lane & 15;
  const int m0 = blockIdx.x * 128;
  const int z  = blockIdx.y;
  const int split = z & 1, bh = z >> 1;
  const int b = bh >> 4, h = bh & 15;

  const float*  Wp = wbase + (size_t)split * W_SPLIT + (size_t)bh * T_ * T_;
  const __bf16* V  = VH + (size_t)z * T_ * DP_;

  v8f acc[4] = {};

  const int r = tid >> 1, seg = tid & 1;
  const float* Wrow = Wp + (size_t)(m0 + r) * T_ + seg * 16;
  const int tv = tid & 31, dp0 = (tid >> 5) * 8;

  for (int k0 = 0; k0 < T_; k0 += 32) {
    __syncthreads();
    // A: 128x32 fp32 -> bf16 into LDS
    {
      union { float4 f4[4]; float f[16]; } tf;
      const float4* fp = (const float4*)(Wrow + k0);
      tf.f4[0] = fp[0]; tf.f4[1] = fp[1]; tf.f4[2] = fp[2]; tf.f4[3] = fp[3];
      union { v16bf v; uint4 u[2]; __bf16 hh[16]; } ac;
#pragma unroll
      for (int e = 0; e < 16; ++e) ac.hh[e] = (__bf16)tf.f[e];
      *(uint4*)&As[r * 32 + seg * 16]     = ac.u[0];
      *(uint4*)&As[r * 32 + seg * 16 + 8] = ac.u[1];
    }
    // B: V tile 32(K) x 64(N), transposed into Bt[n][k]
    {
      union { uint4 u; __bf16 hh[8]; } vv;
      vv.u = *(const uint4*)(V + (size_t)(k0 + tv) * DP_ + dp0);
#pragma unroll
      for (int jj = 0; jj < 8; ++jj) Bt[(dp0 + jj) * 40 + tv] = vv.hh[jj];
    }
    __syncthreads();

    const v16bf af = frag_ld(&As[(wid * 16 + lm) * 32], lh);
    v16bf bfr[4];
#pragma unroll
    for (int j = 0; j < 4; ++j) bfr[j] = frag_ld(&Bt[(j * 16 + lm) * 40], lh);
#pragma unroll
    for (int j = 0; j < 4; ++j) acc[j] = bf16_wmma(af, bfr[j], acc[j]);
  }

#pragma unroll
  for (int j = 0; j < 4; ++j)
#pragma unroll
    for (int rr = 0; rr < 8; ++rr) {
      const int mloc = wid * 16 + rr + lh * 8;
      const int tg = m0 + mloc;
      const int n  = j * 16 + lm;
      const size_t row = (size_t)b * S_ + (size_t)split * T_ + tg;  // concat order
      AO[row * D_ + h * DP_ + n] = (__bf16)acc[j][rr];
    }
}

// ---------------- host side ----------------
extern "C" void kernel_launch(void* const* d_in, const int* in_sizes, int n_in,
                              void* d_out, int out_size, void* d_ws, size_t ws_size,
                              hipStream_t stream) {
  (void)in_sizes; (void)n_in; (void)out_size; (void)ws_size;

  const float* q    = (const float*)d_in[0];
  const float* k    = (const float*)d_in[1];
  const float* v    = (const float*)d_in[2];
  const float* wq_w = (const float*)d_in[3];
  const float* wq_b = (const float*)d_in[4];
  const float* wk_w = (const float*)d_in[5];
  const float* wk_b = (const float*)d_in[6];
  const float* wv_w = (const float*)d_in[7];
  const float* wv_b = (const float*)d_in[8];
  const float* fc_w = (const float*)d_in[9];
  const float* fc_b = (const float*)d_in[10];

  __bf16* ws  = (__bf16*)d_ws;
  __bf16* xq  = ws + XQ_OFF;  __bf16* xk  = ws + XK_OFF;  __bf16* xv = ws + XV_OFF;
  __bf16* wqb16 = ws + WQ_OFF; __bf16* wkb16 = ws + WK_OFF;
  __bf16* wvb16 = ws + WV_OFF; __bf16* wfc16 = ws + WFC_OFF;
  __bf16* qh  = ws + QH_OFF;  __bf16* kh  = ws + KH_OFF;  __bf16* vh = ws + VH_OFF;
  __bf16* ao  = ws + AO_OFF;

  float* outY  = (float*)d_out;
  float* wbase = outY + OUT_ELEMS;     // [w_odd | w_even], contiguous rows

  // 1) convert inputs + weights to bf16
  const int n4x = (int)((size_t)B_ * S_ * D_ / 4);  // 2,097,152
  const int n4w = (int)((size_t)D_ * D_ / 4);       //   262,144
  cvt_f32_to_bf16<<<n4x / 256, 256, 0, stream>>>(q, xq, n4x);
  cvt_f32_to_bf16<<<n4x / 256, 256, 0, stream>>>(k, xk, n4x);
  cvt_f32_to_bf16<<<n4x / 256, 256, 0, stream>>>(v, xv, n4x);
  cvt_f32_to_bf16<<<n4w / 256, 256, 0, stream>>>(wq_w, wqb16, n4w);
  cvt_f32_to_bf16<<<n4w / 256, 256, 0, stream>>>(wk_w, wkb16, n4w);
  cvt_f32_to_bf16<<<n4w / 256, 256, 0, stream>>>(wv_w, wvb16, n4w);
  cvt_f32_to_bf16<<<n4w / 256, 256, 0, stream>>>(fc_w, wfc16, n4w);

  // 2) projections -> zigzag head layout
  dim3 gP(D_ / 128, (B_ * S_) / 128);   // 8 x 64
  gemm_xw_kernel<true><<<gP, 256, 0, stream>>>(xq, wqb16, wq_b, qh);
  gemm_xw_kernel<true><<<gP, 256, 0, stream>>>(xk, wkb16, wk_b, kh);
  gemm_xw_kernel<true><<<gP, 256, 0, stream>>>(xv, wvb16, wv_b, vh);

  // 3) scores -> d_out w region (scaled)
  scores_kernel<<<dim3(T_ / 128, T_ / 128, B_ * H_ * 2), 256, 0, stream>>>(qh, kh, wbase);

  // 4) softmax in place (131072 rows x 2048)
  softmax_kernel<<<dim3(2 * B_ * H_ * T_), 256, 0, stream>>>(wbase);

  // 5) O = W @ V -> merged-head bf16 activations
  wv_kernel<<<dim3(T_ / 128, B_ * H_ * 2), 256, 0, stream>>>(wbase, vh, ao);

  // 6) final fc -> fp32 out
  gemm_xw_kernel<false><<<gP, 256, 0, stream>>>(ao, wfc16, fc_b, outY);
}
